// AttentionWithNoDeconvPos_27986006901448
// MI455X (gfx1250) — matrix operand order repaired and checked by reading, not compiled, not run on hardware
//
#include <hip/hip_runtime.h>
#include <hip/hip_bf16.h>

// MI455X / gfx1250, wave32. WMMA f32_16x16x32_f16 for all GEMMs,
// double-buffered async global->LDS staging (ASYNCcnt), transposed B layouts
// (every fragment = contiguous 32B per lane), bank-conflict-free LDS padding.

typedef __attribute__((ext_vector_type(16))) _Float16 v16h;
typedef __attribute__((ext_vector_type(8)))  float    v8f;

#define B_   16
#define L_   1024
#define D_   512
#define H_   8
#define DH_  64
#define HID_ 4096

// padded LDS strides (in elements) -> lanes spread across all 64 banks
#define TW_  40    // GEMM tile row stride (halfs): 80B
#define QW_  72    // q-tile row stride (halfs): 144B
#define SCW_ 1028  // scores row stride (floats): 4112B
#define PBW_ 1032  // pbuf row stride (halfs): 2064B

static __device__ __forceinline__ int lane_id() { return threadIdx.x & 31; }

static __device__ __forceinline__ v8f wmma_f16(v16h a, v16h b, v8f c) {
  return __builtin_amdgcn_wmma_f32_16x16x32_f16(false, a, false, b, (short)0, c,
                                                false, false);
}

// ---- CDNA5 async global->LDS copy (16B per lane), tracked by ASYNCcnt.
static __device__ __forceinline__ void async_copy_b128(void* lds_dst, const void* gsrc) {
  asm volatile("global_load_async_to_lds_b128 %0, %1, off"
               :: "v"((unsigned)(uintptr_t)lds_dst), "v"(gsrc)
               : "memory");
}
static __device__ __forceinline__ void async_wait0() {
#if __has_builtin(__builtin_amdgcn_s_wait_asynccnt)
  __builtin_amdgcn_s_wait_asynccnt(0);
#else
  asm volatile("s_wait_asynccnt 0" ::: "memory");
#endif
}
static __device__ __forceinline__ void async_wait4() {   // allow 4 in flight
#if __has_builtin(__builtin_amdgcn_s_wait_asynccnt)
  __builtin_amdgcn_s_wait_asynccnt(4);
#else
  asm volatile("s_wait_asynccnt 4" ::: "memory");
#endif
}

// A fragment (16x32 f16), row-major source, `stride` in halfs.
// 16 contiguous halfs (32B) per lane: k = 16*(lane/16) + e.
static __device__ __forceinline__ v16h load_a_frag(const _Float16* base, int stride,
                                                   int m0, int k0) {
  int ln = lane_id();
  const _Float16* p = base + (size_t)(m0 + (ln & 15)) * stride + k0 + ((ln & 16) ? 16 : 0);
  v16h a;
#pragma unroll
  for (int e = 0; e < 16; ++e) a[e] = p[e];
  return a;
}

// B fragment (32x16) from an N-major (transposed) source T[n][k]:
// logical B[k,n] = T[n0+n][k0+k]. 16 contiguous halfs (32B) per lane.
static __device__ __forceinline__ v16h load_bT_frag(const _Float16* T, int n0, int k0,
                                                    int stride) {
  int ln = lane_id();
  const _Float16* p = T + (size_t)(n0 + (ln & 15)) * stride + k0 + ((ln & 16) ? 16 : 0);
  v16h b;
#pragma unroll
  for (int e = 0; e < 16; ++e) b[e] = p[e];
  return b;
}

// ---------------------------------------------------------------- converts
// dst[c*R + r] = (f16)src[r*C + c]  (weights stored N-major for fragment loads)
__global__ void k_conv_transpose(const float* __restrict__ src, _Float16* __restrict__ dst,
                                 int R, int C) {
  int i = blockIdx.x * blockDim.x + threadIdx.x;
  if (i < R * C) {
    int r = i / C, c = i % C;
    dst[(size_t)c * R + r] = (_Float16)src[i];
  }
}

// xe = x[:,:,None]*w_emb + b_emb + pe  -> f16
__global__ void k_embed(const float* __restrict__ x, const float* __restrict__ w_emb,
                        const float* __restrict__ b_emb, const float* __restrict__ pe,
                        _Float16* __restrict__ xe16) {
  int i  = blockIdx.x * blockDim.x + threadIdx.x;   // over B*L*D
  int d  = i & (D_ - 1);
  int bl = i >> 9;
  int l  = bl & (L_ - 1);
  xe16[i] = (_Float16)(x[bl] * w_emb[d] + b_emb[d] + pe[l * D_ + d]);
}

// wv = Wo @ w_un ; c0 = b_un + bo . w_un   (folds the whole Wo GEMM away)
__global__ void k_fold_unembed(const float* __restrict__ Wo, const float* __restrict__ w_un,
                               const float* __restrict__ bo, const float* __restrict__ b_un,
                               float* __restrict__ wv, float* __restrict__ c0) {
  int d = threadIdx.x;
  float acc = 0.f;
  for (int j = 0; j < D_; ++j) acc += Wo[(size_t)d * D_ + j] * w_un[j];
  wv[d] = acc;
  if (d == 0) {
    float c = b_un[0];
    for (int j = 0; j < D_; ++j) c += bo[j] * w_un[j];
    *c0 = c;
  }
}

// ------------------------------------------------- QKV projection (WMMA GEMM)
// Y = xe16 @ W + bias.  Q,K written [B,H,L,DH]; V written transposed [B,H,DH,L].
// Double-buffered async staging: DMA of tile k+1 overlaps WMMA on tile k.
__global__ void __launch_bounds__(128)
k_qkv_gemm(const _Float16* __restrict__ xe16,
           const _Float16* __restrict__ Wq16t, const _Float16* __restrict__ Wk16t,
           const _Float16* __restrict__ Wv16t,
           const float* __restrict__ bq, const float* __restrict__ bk,
           const float* __restrict__ bv,
           _Float16* __restrict__ q16, _Float16* __restrict__ k16,
           _Float16* __restrict__ v16t) {
  __shared__ _Float16 As [2][64 * TW_];
  __shared__ _Float16 Bst[2][64 * TW_];   // N-major tile of W^T

  const int which = blockIdx.z;
  const _Float16* Wt  = (which == 0) ? Wq16t : (which == 1) ? Wk16t : Wv16t;
  const float*   bias = (which == 0) ? bq    : (which == 1) ? bk    : bv;
  _Float16*      outr = (which == 0) ? q16   : k16;   // row-major targets (q,k)

  const int m0   = blockIdx.x * 64;
  const int n0   = blockIdx.y * 64;
  const int tid  = threadIdx.x;
  const int wave = tid >> 5;
  const int ln   = tid & 31;
  const int r0   = tid >> 2;            // 0..31
  const int c0   = (tid & 3) * 8;       // 0,8,16,24

  // exactly 4 async ops per thread per tile (uniform -> count-based wait works)
  auto stage = [&](int buf, int ks) {
    async_copy_b128(&As [buf][(size_t)r0        * TW_ + c0],
                    xe16 + (size_t)(m0 + r0     ) * D_ + ks + c0);
    async_copy_b128(&As [buf][(size_t)(r0 + 32) * TW_ + c0],
                    xe16 + (size_t)(m0 + r0 + 32) * D_ + ks + c0);
    async_copy_b128(&Bst[buf][(size_t)r0        * TW_ + c0],
                    Wt   + (size_t)(n0 + r0     ) * D_ + ks + c0);
    async_copy_b128(&Bst[buf][(size_t)(r0 + 32) * TW_ + c0],
                    Wt   + (size_t)(n0 + r0 + 32) * D_ + ks + c0);
  };

  v8f acc[4] = {};
  stage(0, 0);
  int buf = 0;
  for (int ks = 0; ks < D_; ks += 32, buf ^= 1) {
    if (ks + 32 < D_) { stage(buf ^ 1, ks + 32); async_wait4(); }
    else              { async_wait0(); }
    __syncthreads();

    v16h a = load_a_frag(As[buf], TW_, wave * 16, 0);
#pragma unroll
    for (int nt = 0; nt < 4; ++nt) {
      v16h b = load_bT_frag(Bst[buf], nt * 16, 0, TW_);
      acc[nt] = wmma_f16(a, b, acc[nt]);
    }
    __syncthreads();
  }

#pragma unroll
  for (int nt = 0; nt < 4; ++nt) {
#pragma unroll
    for (int j = 0; j < 8; ++j) {
      int rg = m0 + wave * 16 + j + ((ln & 16) ? 8 : 0);
      int cg = n0 + nt * 16 + (ln & 15);
      _Float16 hv = (_Float16)(acc[nt][j] + bias[cg]);
      int bb = rg >> 10, l = rg & (L_ - 1);
      int h  = cg >> 6,  dh = cg & (DH_ - 1);
      if (which == 2)
        v16t[(((size_t)(bb * H_ + h)) * DH_ + dh) * L_ + l] = hv;   // V^T
      else
        outr[(((size_t)(bb * H_ + h)) * L_ + l) * DH_ + dh] = hv;   // Q, K
    }
  }
}

// -------------------------------------------- fused attention (per 16-row q tile)
// WMMA scores -> gate -> softmax -> atten f32 out (touched once) -> ctx = P@V^T
__global__ void __launch_bounds__(128)
k_attention(const _Float16* __restrict__ q16, const _Float16* __restrict__ k16,
            const _Float16* __restrict__ v16t,
            const float* __restrict__ thr_p, const int* __restrict__ ig_p,
            float* __restrict__ atten_out, _Float16* __restrict__ ctx16) {
  extern __shared__ char smem[];
  const unsigned OFF_PB = 16u * SCW_ * 4u;            // 65792
  const unsigned OFF_QT = OFF_PB + 16u * PBW_ * 2u;   // +33024
  const unsigned OFF_RD = OFF_QT + 16u * QW_ * 2u;    // +2304
  const unsigned OFF_RS = OFF_RD + 512u;
  float*    scores = (float*)smem;
  _Float16* pbuf   = (_Float16*)(smem + OFF_PB);
  _Float16* qtile  = (_Float16*)(smem + OFF_QT);
  float*    red    = (float*)(smem + OFF_RD);
  float*    rsum   = (float*)(smem + OFF_RS);

  const int qt   = blockIdx.x;
  const int bh   = blockIdx.y;
  const int tid  = threadIdx.x;
  const int wave = tid >> 5;
  const int ln   = tid & 31;

  const _Float16* Qbh  = q16  + (size_t)bh * L_ * DH_;
  const _Float16* Kbh  = k16  + (size_t)bh * L_ * DH_;
  const _Float16* Vtbh = v16t + (size_t)bh * DH_ * L_;
  const float thr = *thr_p;
  const int   ig  = *ig_p;

  // async-stage Q tile 16x64 (1 op/thread)
  {
    int r = tid >> 3, c = (tid & 7) * 8;
    async_copy_b128(qtile + r * QW_ + c, Qbh + (size_t)(qt * 16 + r) * DH_ + c);
  }
  async_wait0();
  __syncthreads();

  // ---- phase 1: S = QK^T / 8, gated.  wave w owns keys [256w, 256w+256)
  v16h a0 = load_a_frag(qtile, QW_, 0, 0);
  v16h a1 = load_a_frag(qtile, QW_, 0, 32);
  const float scale = 0.125f;  // 1/sqrt(64)
  for (int kt = 0; kt < 16; ++kt) {
    int key0 = wave * 256 + kt * 16;
    v8f acc = {};
    acc = wmma_f16(a0, load_bT_frag(Kbh, key0, 0,  DH_), acc);
    acc = wmma_f16(a1, load_bT_frag(Kbh, key0, 32, DH_), acc);
#pragma unroll
    for (int j = 0; j < 8; ++j) {
      float s = acc[j] * scale;
      bool keep = ig ? (s > thr) : (s < thr);
      s = keep ? s : -1e9f;
      int r = j + ((ln & 16) ? 8 : 0);
      int c = key0 + (ln & 15);
      scores[r * SCW_ + c] = s;
    }
  }
  __syncthreads();

  // ---- phase 2: softmax (16 rows, 8 threads/row; padded rows -> no bank dup)
  {
    int row = tid >> 3, sub = tid & 7;
    float m = -3.0e38f;
    for (int j = sub; j < L_; j += 8) m = fmaxf(m, scores[row * SCW_ + j]);
    red[row * 8 + sub] = m;
    __syncthreads();
    float mrow = red[row * 8];
#pragma unroll
    for (int t = 1; t < 8; ++t) mrow = fmaxf(mrow, red[row * 8 + t]);
    __syncthreads();
    float s = 0.f;
    for (int j = sub; j < L_; j += 8) {
      float e = __expf(scores[row * SCW_ + j] - mrow);
      scores[row * SCW_ + j] = e;
      s += e;
    }
    red[row * 8 + sub] = s;
    __syncthreads();
    float rs = 0.f;
#pragma unroll
    for (int t = 0; t < 8; ++t) rs += red[row * 8 + t];
    if (sub == 0) rsum[row] = rs;
  }
  __syncthreads();

  // normalize: single coalesced f32 write of atten + f16 copy for P@V
  float* atten_tile = atten_out + (size_t)bh * L_ * L_ + (size_t)(qt * 16) * L_;
  for (int idx = tid; idx < 16 * L_; idx += 128) {
    int row = idx >> 10, col = idx & (L_ - 1);
    float p = scores[row * SCW_ + col] * (1.0f / rsum[row]);
    atten_tile[idx] = p;
    pbuf[row * PBW_ + col] = (_Float16)p;
  }
  __syncthreads();

  // ---- phase 3: ctx = P @ V.  V^T read directly from global (L2-resident,
  // reused by all 64 q-tile blocks of this head); no slab, no barriers.
  v8f cacc = {};
  const int nt = wave;  // output cols [16w,16w+16)
  for (int k0 = 0; k0 < L_; k0 += 32) {
    v16h pa = load_a_frag(pbuf, PBW_, 0, k0);
    v16h vb = load_bT_frag(Vtbh, nt * 16, k0, L_);
    cacc = wmma_f16(pa, vb, cacc);
  }

  {
    int bb = bh >> 3, h = bh & (H_ - 1);
#pragma unroll
    for (int j = 0; j < 8; ++j) {
      int r = qt * 16 + j + ((ln & 16) ? 8 : 0);
      int c = h * DH_ + nt * 16 + (ln & 15);
      ctx16[((size_t)(bb * L_ + r)) * D_ + c] = (_Float16)cacc[j];
    }
  }
}

// x_atten = ctx @ wv + c0   (Wo GEMM folded into wv)
__global__ void __launch_bounds__(256)
k_unembed(const _Float16* __restrict__ ctx16, const float* __restrict__ wv,
          const float* __restrict__ c0, float* __restrict__ xatt_f32,
          _Float16* __restrict__ xatt16) {
  int wave = threadIdx.x >> 5, ln = threadIdx.x & 31;
  int row = blockIdx.x * 8 + wave;
  const _Float16* p = ctx16 + (size_t)row * D_;
  float acc = 0.f;
  for (int j = ln; j < D_; j += 32) acc += (float)p[j] * wv[j];
#pragma unroll
  for (int off = 16; off >= 1; off >>= 1) acc += __shfl_xor(acc, off, 32);
  if (ln == 0) {
    float v = acc + *c0;
    xatt_f32[row] = v;
    xatt16[row]   = (_Float16)v;
  }
}

// xf = x_atten @ W_ff1 + b_ff1   (M=16 -> one WMMA tile row)
__global__ void __launch_bounds__(128)
k_ff1(const _Float16* __restrict__ xatt16, const _Float16* __restrict__ Wt,
      const float* __restrict__ b_ff1, float* __restrict__ out) {
  __shared__ _Float16 As [16 * TW_];
  __shared__ _Float16 Bst[64 * TW_];
  const int n0 = blockIdx.x * 64;
  const int tid = threadIdx.x, wave = tid >> 5, ln = tid & 31;
  const int r0 = tid >> 2, c0 = (tid & 3) * 8;
  v8f acc = {};
  for (int ks = 0; ks < L_; ks += 32) {
    __syncthreads();
    if (tid < 64)
      async_copy_b128(&As[(size_t)r0 * TW_ + c0],
                      xatt16 + (size_t)r0 * L_ + ks + c0);
    async_copy_b128(&Bst[(size_t)r0        * TW_ + c0],
                    Wt + (size_t)(n0 + r0     ) * L_ + ks + c0);
    async_copy_b128(&Bst[(size_t)(r0 + 32) * TW_ + c0],
                    Wt + (size_t)(n0 + r0 + 32) * L_ + ks + c0);
    async_wait0();
    __syncthreads();
    v16h a = load_a_frag(As, TW_, 0, 0);
    v16h b = load_bT_frag(Bst, wave * 16, 0, TW_);
    acc = wmma_f16(a, b, acc);
  }
#pragma unroll
  for (int j = 0; j < 8; ++j) {
    int r = j + ((ln & 16) ? 8 : 0);
    int c = n0 + wave * 16 + (ln & 15);
    out[(size_t)r * HID_ + c] = acc[j] + b_ff1[c];
  }
}

extern "C" void kernel_launch(void* const* d_in, const int* in_sizes, int n_in,
                              void* d_out, int out_size, void* d_ws, size_t ws_size,
                              hipStream_t stream) {
  (void)in_sizes; (void)n_in; (void)out_size; (void)ws_size;
  const float* x     = (const float*)d_in[0];
  const float* thr   = (const float*)d_in[1];
  const float* w_emb = (const float*)d_in[2];
  const float* b_emb = (const float*)d_in[3];
  const float* pe    = (const float*)d_in[4];
  const float* Wq    = (const float*)d_in[5];
  const float* bq    = (const float*)d_in[6];
  const float* Wk    = (const float*)d_in[7];
  const float* bk    = (const float*)d_in[8];
  const float* Wv    = (const float*)d_in[9];
  const float* bv    = (const float*)d_in[10];
  const float* Wo    = (const float*)d_in[11];
  const float* bo    = (const float*)d_in[12];
  const float* w_un  = (const float*)d_in[13];
  const float* b_un  = (const float*)d_in[14];
  const float* Wff1  = (const float*)d_in[15];
  const float* bff1  = (const float*)d_in[16];
  const int*   isg   = (const int*)d_in[17];

  char* ws = (char*)d_ws;
  size_t off = 0;
  const size_t SZ_BLD = (size_t)B_ * L_ * D_ * sizeof(_Float16);  // 16 MB
  _Float16* xe16  = (_Float16*)(ws + off); off += SZ_BLD;
  _Float16* q16   = (_Float16*)(ws + off); off += SZ_BLD;
  _Float16* k16   = (_Float16*)(ws + off); off += SZ_BLD;
  _Float16* v16t  = (_Float16*)(ws + off); off += SZ_BLD;
  _Float16* ctx16 = (_Float16*)(ws + off); off += SZ_BLD;
  _Float16* Wq16t = (_Float16*)(ws + off); off += (size_t)D_ * D_ * 2;
  _Float16* Wk16t = (_Float16*)(ws + off); off += (size_t)D_ * D_ * 2;
  _Float16* Wv16t = (_Float16*)(ws + off); off += (size_t)D_ * D_ * 2;
  _Float16* Wf16t = (_Float16*)(ws + off); off += (size_t)L_ * HID_ * 2;
  _Float16* xat16 = (_Float16*)(ws + off); off += (size_t)B_ * L_ * 2;
  float*    wvv   = (float*)(ws + off);    off += D_ * sizeof(float);
  float*    c0    = (float*)(ws + off);    off += 256;

  // output layout: x_img [16*4096] | atten [16*8*1024*1024] | x_atten [16*1024]
  float* out_ximg  = (float*)d_out;
  float* out_atten = (float*)d_out + (size_t)B_ * HID_;
  float* out_xatt  = out_atten + (size_t)B_ * H_ * L_ * L_;

  const int DD = D_ * D_, LH = L_ * HID_;
  k_conv_transpose<<<(DD + 255) / 256, 256, 0, stream>>>(Wq, Wq16t, D_, D_);
  k_conv_transpose<<<(DD + 255) / 256, 256, 0, stream>>>(Wk, Wk16t, D_, D_);
  k_conv_transpose<<<(DD + 255) / 256, 256, 0, stream>>>(Wv, Wv16t, D_, D_);
  k_conv_transpose<<<(LH + 255) / 256, 256, 0, stream>>>(Wff1, Wf16t, L_, HID_);

  k_embed<<<(B_ * L_ * D_) / 256, 256, 0, stream>>>(x, w_emb, b_emb, pe, xe16);
  k_fold_unembed<<<1, D_, 0, stream>>>(Wo, w_un, bo, b_un, wvv, c0);

  dim3 gq(B_ * L_ / 64, D_ / 64, 3);
  k_qkv_gemm<<<gq, 128, 0, stream>>>(xe16, Wq16t, Wk16t, Wv16t, bq, bk, bv,
                                     q16, k16, v16t);

  dim3 ga(L_ / 16, B_ * H_);
  size_t smem = (size_t)16 * SCW_ * 4 + (size_t)16 * PBW_ * 2
              + (size_t)16 * QW_ * 2 + 512 + 64;
  k_attention<<<ga, 128, smem, stream>>>(q16, k16, v16t, thr, isg, out_atten, ctx16);

  k_unembed<<<(B_ * L_) / 8, 256, 0, stream>>>(ctx16, wvv, c0, out_xatt, xat16);

  k_ff1<<<HID_ / 64, 128, 0, stream>>>(xat16, Wf16t, bff1, out_ximg);
}
